// GatedDeltaNet_80934363726541
// MI455X (gfx1250) — matrix operand order repaired and checked by reading, not compiled, not run on hardware
//
#include <hip/hip_runtime.h>
#include <hip/hip_bf16.h>

typedef unsigned short u16;
typedef __attribute__((ext_vector_type(16))) __bf16 v16bf;
typedef __attribute__((ext_vector_type(8)))  float  v8f;

#define WMMA_BF16(a, b, c) \
  __builtin_amdgcn_wmma_f32_16x16x32_bf16(false, (a), false, (b), (short)0, (c), false, false)

// ---------------- problem dims ----------------
#define SS        2048
#define HIDN      2048
#define CONV_DIM  8192
#define VAL_DIM   4096
#define KEY_DIM   2048
#define BS_TOT    4096   // B*S
#define CHUNK     64
#define NCHUNK    32

// ---------------- LDS pitches for delta kernel ----------------
#define PH    136   // u16 pitch, 128-wide tiles
#define PF    132   // f32 pitch, 128-wide tiles
#define P64F  68    // f32 pitch, 64-wide tiles
#define P64H  72    // u16 pitch, 64-wide tiles

static constexpr size_t DELTA_LDS_BYTES =
    (size_t)(128 * PH + 6 * 64 * PH + 64 * P64H) * 2 +           // bf16 tiles
    (size_t)(2 * 64 * PF + 2 * 64 * P64F + 3 * 64 + 256) * 4;    // f32 tiles + scalars

// ---------------- bf16 helpers ----------------
__device__ __forceinline__ u16 f2bf_u(float f) {
  union { float f; unsigned int u; } c; c.f = f;
  unsigned int u = c.u + 0x7FFFu + ((c.u >> 16) & 1u);
  return (u16)(u >> 16);
}
__device__ __forceinline__ float bf2f(u16 h) {
  union { unsigned int u; float f; } c; c.u = ((unsigned int)h) << 16;
  return c.f;
}
__device__ __forceinline__ __bf16 us2bf(u16 h) {
  union { u16 u; __bf16 b; } c; c.u = h; return c.b;
}
__device__ __forceinline__ __bf16 f2bf(float f) { return us2bf(f2bf_u(f)); }

// ---------------- WMMA fragment loaders (CDNA5 wave32 layouts) ----------------
// A fragment: A[m][k], source row-major with leading dim ld (u16/bf16 elements).
// lane L: row = L&15; elements k = (L>>4)*16 + e.
__device__ __forceinline__ v16bf ldA_rm(const u16* p, int ld, int lane) {
  const u16* q0 = p + (lane & 15) * ld + ((lane >> 4) << 4);
  v16bf r;
#pragma unroll
  for (int e = 0; e < 8; ++e) { r[e] = us2bf(q0[e]); r[e + 8] = us2bf(q0[e + 8]); }
  return r;
}
// A fragment transposed: A[m][k] = X[k][m], X row-major [K][*] with leading dim ld.
__device__ __forceinline__ v16bf ldA_tr(const u16* p, int ld, int lane) {
  int m = lane & 15, kh = (lane >> 4) << 4;
  const u16* q0 = p + kh * ld + m;
  const u16* q1 = p + (kh + 8) * ld + m;
  v16bf r;
#pragma unroll
  for (int e = 0; e < 8; ++e) { r[e] = us2bf(q0[e * ld]); r[e + 8] = us2bf(q1[e * ld]); }
  return r;
}
// B fragment for C = A·Bsrcᵀ: Bsrc[n][k] row-major (K contiguous).
// lane L: col n = L&15; elements k = (L>>4)*16 + e.
__device__ __forceinline__ v16bf ldB_nt(const u16* p, int ld, int lane) {
  const u16* q0 = p + (lane & 15) * ld + ((lane >> 4) << 4);
  v16bf r;
#pragma unroll
  for (int e = 0; e < 16; ++e) r[e] = us2bf(q0[e]);
  return r;
}
// B fragment for C = A·Bsrc: Bsrc[k][n] row-major (N contiguous), u16 source.
__device__ __forceinline__ v16bf ldB_nn_u16(const u16* p, int ld, int lane) {
  const u16* q0 = p + (((lane >> 4) << 4)) * ld + (lane & 15);
  v16bf r;
#pragma unroll
  for (int e = 0; e < 16; ++e) r[e] = us2bf(q0[e * ld]);
  return r;
}
// Same, fp32 source converted on the fly.
__device__ __forceinline__ v16bf ldB_nn_f32(const float* p, int ld, int lane) {
  const float* q0 = p + (((lane >> 4) << 4)) * ld + (lane & 15);
  v16bf r;
#pragma unroll
  for (int e = 0; e < 16; ++e) r[e] = f2bf(q0[e * ld]);
  return r;
}
// C fragment store: element i -> row m = i + (lane>>4)*8, col n = lane&15.
__device__ __forceinline__ void stC_f32(float* p, int ld, int lane, v8f c) {
  int n = lane & 15, mo = (lane >> 4) << 3;
#pragma unroll
  for (int i = 0; i < 8; ++i) p[(mo + i) * ld + n] = c[i];
}
__device__ __forceinline__ void stC_u16(u16* p, int ld, int lane, v8f c) {
  int n = lane & 15, mo = (lane >> 4) << 3;
#pragma unroll
  for (int i = 0; i < 8; ++i) p[(mo + i) * ld + n] = f2bf_u(c[i]);
}

// ================= kernel 1: fp32 -> bf16 conversion =================
__global__ __launch_bounds__(256) void cvt_bf16_k(const float* __restrict__ in,
                                                  u16* __restrict__ out, int n) {
  int i = blockIdx.x * 256 + threadIdx.x;
  int stride = gridDim.x * 256;
  for (; i < n; i += stride) out[i] = f2bf_u(in[i]);
}

// ================= kernel 2: bf16 GEMM, C[M,N] = A[M,K] · B[N,K]^T ==========
// block = 256 threads (8 waves); block tile 256(M) x 64(N); wave tile 32x64.
// 8 WMMAs per k-step from 6 fragment loads (~21 FLOP/B from cache).
__global__ __launch_bounds__(256) void gemm_bf16_nt(const u16* __restrict__ A,
                                                    const u16* __restrict__ B,
                                                    float* __restrict__ C,
                                                    int M, int N, int K) {
  const int wave = threadIdx.x >> 5, lane = threadIdx.x & 31;
  const int row0 = (blockIdx.x * 8 + wave) * 32;
  const int col0 = blockIdx.y * 64;
  v8f acc0[4], acc1[4];
#pragma unroll
  for (int t = 0; t < 4; ++t) { acc0[t] = (v8f){}; acc1[t] = (v8f){}; }
  const u16* Ap0 = A + (size_t)row0 * K;
  const u16* Ap1 = A + (size_t)(row0 + 16) * K;
  const u16* Bp0 = B + (size_t)(col0 +  0) * K;
  const u16* Bp1 = B + (size_t)(col0 + 16) * K;
  const u16* Bp2 = B + (size_t)(col0 + 32) * K;
  const u16* Bp3 = B + (size_t)(col0 + 48) * K;
  for (int k = 0; k < K; k += 32) {
    // prefetch next k-step lines (global_prefetch path)
    if (k + 64 < K) {
      __builtin_prefetch(Ap0 + k + 64, 0, 1);
      __builtin_prefetch(Ap1 + k + 64, 0, 1);
      __builtin_prefetch(Bp0 + k + 64, 0, 1);
      __builtin_prefetch(Bp2 + k + 64, 0, 1);
    }
    v16bf a0 = ldA_rm(Ap0 + k, K, lane);
    v16bf a1 = ldA_rm(Ap1 + k, K, lane);
    v16bf b0 = ldB_nt(Bp0 + k, K, lane);
    v16bf b1 = ldB_nt(Bp1 + k, K, lane);
    v16bf b2 = ldB_nt(Bp2 + k, K, lane);
    v16bf b3 = ldB_nt(Bp3 + k, K, lane);
    acc0[0] = WMMA_BF16(a0, b0, acc0[0]);
    acc0[1] = WMMA_BF16(a0, b1, acc0[1]);
    acc0[2] = WMMA_BF16(a0, b2, acc0[2]);
    acc0[3] = WMMA_BF16(a0, b3, acc0[3]);
    acc1[0] = WMMA_BF16(a1, b0, acc1[0]);
    acc1[1] = WMMA_BF16(a1, b1, acc1[1]);
    acc1[2] = WMMA_BF16(a1, b2, acc1[2]);
    acc1[3] = WMMA_BF16(a1, b3, acc1[3]);
  }
  int n = lane & 15, mo = (lane >> 4) << 3;
#pragma unroll
  for (int i = 0; i < 8; ++i) {
    size_t r0 = (size_t)(row0 + mo + i) * N + col0 + n;
    size_t r1 = (size_t)(row0 + 16 + mo + i) * N + col0 + n;
#pragma unroll
    for (int t = 0; t < 4; ++t) {
      C[r0 + 16 * t] = acc0[t][i];
      C[r1 + 16 * t] = acc1[t][i];
    }
  }
}

// ================= kernel 3: b/a projections + gates =================
// one block per (b,s); beta = sigmoid(x·W_b), g = -exp(A_log)*softplus(x·W_a+dt_bias)
// 4 threads per output dot product.
__global__ __launch_bounds__(256) void ba_proj_k(const float* __restrict__ x,
                                                 const float* __restrict__ Wb,
                                                 const float* __restrict__ Wa,
                                                 const float* __restrict__ dt_bias,
                                                 const float* __restrict__ A_log,
                                                 float* __restrict__ beta,
                                                 float* __restrict__ g) {
  __shared__ float xs[HIDN];
  __shared__ float pr[256];
  const int bs = blockIdx.x;
  const float* xr = x + (size_t)bs * HIDN;
  for (int i = threadIdx.x; i < HIDN; i += 256) xs[i] = xr[i];
  __syncthreads();
  {
    int o = threadIdx.x >> 2, p = threadIdx.x & 3;
    const float* wr = ((o < 32) ? Wb : Wa) + (size_t)(o & 31) * HIDN;
    float acc = 0.f;
#pragma unroll 4
    for (int i = p * 512; i < p * 512 + 512; ++i) acc += xs[i] * wr[i];
    pr[threadIdx.x] = acc;
  }
  __syncthreads();
  if (threadIdx.x < 64) {
    int o = threadIdx.x, n = o & 31;
    float acc = pr[o * 4] + pr[o * 4 + 1] + pr[o * 4 + 2] + pr[o * 4 + 3];
    if (o < 32) {
      beta[(size_t)bs * 32 + n] = 1.f / (1.f + __expf(-acc));
    } else {
      float xg = acc + dt_bias[n];
      float sp = (xg > 20.f) ? xg : log1pf(__expf(xg));
      g[(size_t)bs * 32 + n] = -__expf(A_log[n]) * sp;
    }
  }
}

// ================= kernel 4: causal depthwise conv + SiLU + split/L2-norm =====
__global__ __launch_bounds__(256) void conv_silu_split_k(const float* __restrict__ mixed,
                                                         const float* __restrict__ conv_w,
                                                         float* __restrict__ qn,
                                                         float* __restrict__ kn,
                                                         float* __restrict__ vg) {
  __shared__ float buf[CONV_DIM];
  __shared__ float part[256];
  __shared__ float rinv[32];
  const int bs = blockIdx.x;
  const int s = bs & (SS - 1);
  const float* base = mixed + (size_t)bs * CONV_DIM;
  for (int c = threadIdx.x; c < CONV_DIM; c += 256) {
    float acc = 0.f;
#pragma unroll
    for (int kk = 0; kk < 4; ++kk) {
      int sp = s + kk - 3;
      if (sp >= 0) acc += base[(long)(kk - 3) * CONV_DIM + c] * conv_w[c * 4 + kk];
    }
    buf[c] = acc / (1.f + __expf(-acc));   // SiLU
  }
  __syncthreads();
  // L2 norms over 32 heads of 128 (16 q heads then 16 k heads = channels [0,4096))
  {
    int grp = threadIdx.x >> 3, jj = threadIdx.x & 7;
    float ss = 0.f;
#pragma unroll
    for (int e = 0; e < 16; ++e) { float t = buf[grp * 128 + jj * 16 + e]; ss += t * t; }
    part[threadIdx.x] = ss;
  }
  __syncthreads();
  if (threadIdx.x < 32) {
    float s2 = 0.f;
#pragma unroll
    for (int e = 0; e < 8; ++e) s2 += part[threadIdx.x * 8 + e];
    rinv[threadIdx.x] = 1.f / fmaxf(sqrtf(s2), 1e-12f);
  }
  __syncthreads();
  const float qscale = 0.08838834764831845f;  // DK^-0.5
  for (int c = threadIdx.x; c < CONV_DIM; c += 256) {
    float val = buf[c];
    if (c < KEY_DIM)            qn[(size_t)bs * KEY_DIM + c] = val * rinv[c >> 7] * qscale;
    else if (c < 2 * KEY_DIM)   kn[(size_t)bs * KEY_DIM + (c - KEY_DIM)] = val * rinv[c >> 7];
    else                        vg[(size_t)bs * VAL_DIM + (c - 2 * KEY_DIM)] = val;
  }
}

// ================= kernel 5: chunked gated delta rule + fused RMSNorm gate =====
// one block per (b, head); 8 waves; state S[128][128] in WMMA C-fragments (wave w
// owns d-rows [16w,16w+16), 8 n-tiles) + bf16 shadow in LDS for use as B operand.
__global__ __launch_bounds__(256) void delta_scan_k(const float* __restrict__ qn,
                                                    const float* __restrict__ kn,
                                                    const float* __restrict__ vg,
                                                    const float* __restrict__ beta_g,
                                                    const float* __restrict__ g_g,
                                                    const float* __restrict__ zbuf,
                                                    const float* __restrict__ norm_w,
                                                    u16* __restrict__ normed_bf) {
  extern __shared__ char smem[];
  u16* Sbf  = (u16*)smem;                 // [128][PH]  bf16 shadow of state
  u16* qbf  = Sbf  + 128 * PH;            // [64][PH]   q
  u16* qgbf = qbf  + 64 * PH;             // q * e^gc
  u16* kbf  = qgbf + 64 * PH;             // k
  u16* kxbf = kbf  + 64 * PH;             // k*beta, later k*e^(gl-gc)
  u16* kgbf = kxbf + 64 * PH;             // k*beta*e^gc
  u16* wbf  = kgbf + 64 * PH;             // kcd (T·(k_b e^gc)) bf16
  u16* mbf  = wbf  + 64 * PH;             // [64][P64H]  T bf16, later attn bf16
  float* vbuf = (float*)(mbf + 64 * P64H);// [64][PF]  v*beta, later v_new
  float* ubuf = vbuf + 64 * PF;           // [64][PF]  u, later out
  float* Abuf = ubuf + 64 * PF;           // [64][P64F] A / attn fp32
  float* Tbuf = Abuf + 64 * P64F;         // [64][P64F] T fp32
  float* gcs  = Tbuf + 64 * P64F;         // [64] cumsum g
  float* bets = gcs + 64;                 // [64] beta
  float* rstd = bets + 64;                // [64] rms scale
  float* red  = rstd + 64;                // [256] scratch reductions

  const int tid = threadIdx.x;
  const int wave = tid >> 5, lane = tid & 31;
  const int bb = blockIdx.x >> 5, hh = blockIdx.x & 31, hq = hh >> 1;

  v8f Sacc[8];
#pragma unroll
  for (int t = 0; t < 8; ++t) Sacc[t] = (v8f){};
  for (int i = tid; i < 128 * PH; i += 256) Sbf[i] = 0;
  __syncthreads();

  for (int nc = 0; nc < NCHUNK; ++nc) {
    const int s0 = nc * CHUNK;
    // ---- gates: load beta/g, cumsum g within chunk ----
    if (tid < 64) {
      size_t sr = (size_t)(bb * SS + s0 + tid) * 32 + hh;
      bets[tid] = beta_g[sr];
      red[tid]  = g_g[sr];
    }
    __syncthreads();
    if (tid == 0) {
      float run = 0.f;
      for (int i = 0; i < 64; ++i) { run += red[i]; gcs[i] = run; }
    }
    __syncthreads();
    const float gl = gcs[63];
    // ---- build chunk tiles ----
    for (int idx = tid; idx < 64 * 128; idx += 256) {
      int r = idx >> 7, d = idx & 127;
      size_t srow = (size_t)(bb * SS + s0 + r);
      float qv = qn[srow * KEY_DIM + hq * 128 + d];
      float kv = kn[srow * KEY_DIM + hq * 128 + d];
      float vv = vg[srow * VAL_DIM + hh * 128 + d];
      float eg = __expf(gcs[r]);
      float be = bets[r];
      qbf [r * PH + d] = f2bf_u(qv);
      qgbf[r * PH + d] = f2bf_u(qv * eg);
      kbf [r * PH + d] = f2bf_u(kv);
      kxbf[r * PH + d] = f2bf_u(kv * be);
      kgbf[r * PH + d] = f2bf_u(kv * be * eg);
      vbuf[r * PF + d] = vv * be;
    }
    __syncthreads();
    // ---- A_raw = (k*beta)·kᵀ  [64x64], 16 tiles, 2 per wave ----
    for (int t = wave * 2; t < wave * 2 + 2; ++t) {
      int ti = t >> 2, tj = t & 3;
      v8f acc = {};
      for (int k = 0; k < 128; k += 32) {
        v16bf a = ldA_rm(kxbf + ti * 16 * PH + k, PH, lane);
        v16bf b = ldB_nt(kbf + tj * 16 * PH + k, PH, lane);
        acc = WMMA_BF16(a, b, acc);
      }
      stC_f32(Abuf + ti * 16 * P64F + tj * 16, P64F, lane, acc);
    }
    __syncthreads();
    // decay + strict mask + negate
    for (int idx = tid; idx < 64 * 64; idx += 256) {
      int i = idx >> 6, j = idx & 63;
      float v = (j < i) ? -Abuf[i * P64F + j] * __expf(gcs[i] - gcs[j]) : 0.f;
      Abuf[i * P64F + j] = v;
    }
    __syncthreads();
    // ---- T = (I-A)^{-1}  forward substitution (row serial, 64 cols parallel) ----
    for (int i = 0; i < 64; ++i) {
      if (tid < 64) {
        float acc = (tid == i) ? 1.f : 0.f;
        for (int l = 0; l < i; ++l) acc += Abuf[i * P64F + l] * Tbuf[l * P64F + tid];
        Tbuf[i * P64F + tid] = acc;
      }
      __syncthreads();
    }
    for (int idx = tid; idx < 64 * 64; idx += 256)
      mbf[(idx >> 6) * P64H + (idx & 63)] = f2bf_u(Tbuf[(idx >> 6) * P64F + (idx & 63)]);
    __syncthreads();
    // ---- u = T·(v*beta), kcd = T·(k*beta*e^gc)  [64x128], 32 tiles, 4/wave ----
    for (int t = wave * 4; t < wave * 4 + 4; ++t) {
      int ti = t >> 3, tj = t & 7;
      v8f accU = {}, accW = {};
      for (int k = 0; k < 64; k += 32) {
        v16bf a  = ldA_rm(mbf + ti * 16 * P64H + k, P64H, lane);
        v16bf bu = ldB_nn_f32(vbuf + k * PF + tj * 16, PF, lane);
        v16bf bw = ldB_nn_u16(kgbf + k * PH + tj * 16, PH, lane);
        accU = WMMA_BF16(a, bu, accU);
        accW = WMMA_BF16(a, bw, accW);
      }
      stC_f32(ubuf + ti * 16 * PF + tj * 16, PF, lane, accU);
      stC_u16(wbf + ti * 16 * PH + tj * 16, PH, lane, accW);
    }
    __syncthreads();
    // ---- attn = (q·kᵀ) * decay(incl diag)  -> bf16 in mbf ----
    for (int t = wave * 2; t < wave * 2 + 2; ++t) {
      int ti = t >> 2, tj = t & 3;
      v8f acc = {};
      for (int k = 0; k < 128; k += 32) {
        v16bf a = ldA_rm(qbf + ti * 16 * PH + k, PH, lane);
        v16bf b = ldB_nt(kbf + tj * 16 * PH + k, PH, lane);
        acc = WMMA_BF16(a, b, acc);
      }
      stC_f32(Abuf + ti * 16 * P64F + tj * 16, P64F, lane, acc);
    }
    __syncthreads();
    for (int idx = tid; idx < 64 * 64; idx += 256) {
      int i = idx >> 6, j = idx & 63;
      float v = (j <= i) ? Abuf[i * P64F + j] * __expf(gcs[i] - gcs[j]) : 0.f;
      mbf[i * P64H + j] = f2bf_u(v);
    }
    __syncthreads();
    // ---- v_new = u - kcd·S_prev ----
    for (int t = wave * 4; t < wave * 4 + 4; ++t) {
      int ti = t >> 3, tj = t & 7;
      v8f acc = {};
      for (int k = 0; k < 128; k += 32) {
        v16bf a = ldA_rm(wbf + ti * 16 * PH + k, PH, lane);
        v16bf b = ldB_nn_u16(Sbf + k * PH + tj * 16, PH, lane);
        acc = WMMA_BF16(a, b, acc);
      }
      int n = lane & 15, mo = (lane >> 4) << 3;
#pragma unroll
      for (int i = 0; i < 8; ++i) {
        int m = ti * 16 + mo + i, c = tj * 16 + n;
        vbuf[m * PF + c] = ubuf[m * PF + c] - acc[i];
      }
    }
    __syncthreads();
    // ---- out = (q e^gc)·S_prev + attn·v_new  -> ubuf ----
    for (int t = wave * 4; t < wave * 4 + 4; ++t) {
      int ti = t >> 3, tj = t & 7;
      v8f acc = {};
      for (int k = 0; k < 128; k += 32) {
        v16bf a = ldA_rm(qgbf + ti * 16 * PH + k, PH, lane);
        v16bf b = ldB_nn_u16(Sbf + k * PH + tj * 16, PH, lane);
        acc = WMMA_BF16(a, b, acc);
      }
      for (int k = 0; k < 64; k += 32) {
        v16bf a = ldA_rm(mbf + ti * 16 * P64H + k, P64H, lane);
        v16bf b = ldB_nn_f32(vbuf + k * PF + tj * 16, PF, lane);
        acc = WMMA_BF16(a, b, acc);
      }
      stC_f32(ubuf + ti * 16 * PF + tj * 16, PF, lane, acc);
    }
    __syncthreads();
    // ---- k·e^(gl-gc) for state update ----
    for (int idx = tid; idx < 64 * 128; idx += 256) {
      int r = idx >> 7, d = idx & 127;
      kxbf[r * PH + d] = f2bf_u(bf2f(kbf[r * PH + d]) * __expf(gl - gcs[r]));
    }
    __syncthreads();
    // ---- S = S*e^gl + (k e^(gl-gc))ᵀ · v_new  (accumulate into fragments) ----
    {
      float egl = __expf(gl);
#pragma unroll
      for (int t = 0; t < 8; ++t)
#pragma unroll
        for (int i = 0; i < 8; ++i) Sacc[t][i] *= egl;
      for (int t = 0; t < 8; ++t) {
        for (int k = 0; k < 64; k += 32) {
          v16bf a = ldA_tr(kxbf + k * PH + wave * 16, PH, lane);
          v16bf b = ldB_nn_f32(vbuf + k * PF + t * 16, PF, lane);
          Sacc[t] = WMMA_BF16(a, b, Sacc[t]);
        }
      }
      // refresh bf16 shadow (disjoint rows per wave; consumed only next chunk)
      int n = lane & 15, mo = (lane >> 4) << 3;
#pragma unroll
      for (int t = 0; t < 8; ++t)
#pragma unroll
        for (int i = 0; i < 8; ++i)
          Sbf[(wave * 16 + mo + i) * PH + t * 16 + n] = f2bf_u(Sacc[t][i]);
    }
    // ---- fused RMSNorm + SiLU(z) gate, bf16 store ----
    {
      int rr = tid >> 2, pp = tid & 3;
      float ss = 0.f;
#pragma unroll
      for (int e = 0; e < 32; ++e) { float v = ubuf[rr * PF + pp * 32 + e]; ss += v * v; }
      red[tid] = ss;
      __syncthreads();
      if (tid < 64) {
        float s2 = red[tid * 4] + red[tid * 4 + 1] + red[tid * 4 + 2] + red[tid * 4 + 3];
        rstd[tid] = rsqrtf(s2 * (1.f / 128.f) + 1e-6f);
      }
      __syncthreads();
      for (int idx = tid; idx < 64 * 128; idx += 256) {
        int r = idx >> 7, d = idx & 127;
        size_t srow = (size_t)(bb * SS + s0 + r);
        float val = ubuf[r * PF + d] * rstd[r] * norm_w[d];
        float zv = zbuf[srow * VAL_DIM + hh * 128 + d];
        val *= zv / (1.f + __expf(-zv));
        normed_bf[srow * VAL_DIM + hh * 128 + d] = f2bf_u(val);
      }
    }
    __syncthreads();
  }
}

// =========================== host launcher ===========================
extern "C" void kernel_launch(void* const* d_in, const int* in_sizes, int n_in,
                              void* d_out, int out_size, void* d_ws, size_t ws_size,
                              hipStream_t stream) {
  (void)in_sizes; (void)n_in; (void)out_size; (void)ws_size;
  const float* x      = (const float*)d_in[0];
  const float* W_qkv  = (const float*)d_in[1];
  const float* W_z    = (const float*)d_in[2];
  const float* W_b    = (const float*)d_in[3];
  const float* W_a    = (const float*)d_in[4];
  const float* W_out  = (const float*)d_in[5];
  const float* conv_w = (const float*)d_in[6];
  const float* dt_b   = (const float*)d_in[7];
  const float* A_log  = (const float*)d_in[8];
  const float* norm_w = (const float*)d_in[9];
  float* out = (float*)d_out;

  char* ws = (char*)d_ws;
  size_t off = 0;
  auto carve = [&](size_t bytes) {
    char* p = ws + off;
    off += (bytes + 255) & ~(size_t)255;
    return p;
  };
  u16*   xbf    = (u16*)  carve((size_t)BS_TOT * HIDN * 2);        // x bf16
  u16*   wqkvbf = (u16*)  carve((size_t)CONV_DIM * HIDN * 2);      // W_qkv bf16
  u16*   wzbf   = (u16*)  carve((size_t)VAL_DIM * HIDN * 2);       // W_z bf16
  u16*   woutbf = (u16*)  carve((size_t)HIDN * VAL_DIM * 2);       // W_out bf16
  float* mixed  = (float*)carve((size_t)BS_TOT * CONV_DIM * 4);
  float* zb     = (float*)carve((size_t)BS_TOT * VAL_DIM * 4);
  float* betab  = (float*)carve((size_t)BS_TOT * 32 * 4);
  float* gb     = (float*)carve((size_t)BS_TOT * 32 * 4);
  float* qnb    = (float*)carve((size_t)BS_TOT * KEY_DIM * 4);
  float* knb    = (float*)carve((size_t)BS_TOT * KEY_DIM * 4);
  float* vb     = (float*)carve((size_t)BS_TOT * VAL_DIM * 4);
  u16*   nrm    = (u16*)  carve((size_t)BS_TOT * VAL_DIM * 2);

  cvt_bf16_k<<<4096, 256, 0, stream>>>(x, xbf, BS_TOT * HIDN);
  cvt_bf16_k<<<4096, 256, 0, stream>>>(W_qkv, wqkvbf, CONV_DIM * HIDN);
  cvt_bf16_k<<<4096, 256, 0, stream>>>(W_z, wzbf, VAL_DIM * HIDN);
  cvt_bf16_k<<<4096, 256, 0, stream>>>(W_out, woutbf, HIDN * VAL_DIM);

  // mixed = x · W_qkvᵀ   [4096 x 8192]
  gemm_bf16_nt<<<dim3(BS_TOT / 256, CONV_DIM / 64), 256, 0, stream>>>(
      xbf, wqkvbf, mixed, BS_TOT, CONV_DIM, HIDN);
  // z = x · W_zᵀ         [4096 x 4096]
  gemm_bf16_nt<<<dim3(BS_TOT / 256, VAL_DIM / 64), 256, 0, stream>>>(
      xbf, wzbf, zb, BS_TOT, VAL_DIM, HIDN);

  ba_proj_k<<<BS_TOT, 256, 0, stream>>>(x, W_b, W_a, dt_b, A_log, betab, gb);
  conv_silu_split_k<<<BS_TOT, 256, 0, stream>>>(mixed, conv_w, qnb, knb, vb);

  delta_scan_k<<<64, 256, DELTA_LDS_BYTES, stream>>>(qnb, knb, vb, betab, gb, zb,
                                                     norm_w, nrm);

  // out = normed · W_outᵀ   [4096 x 2048]
  gemm_bf16_nt<<<dim3(BS_TOT / 256, HIDN / 64), 256, 0, stream>>>(
      nrm, woutbf, out, BS_TOT, HIDN, VAL_DIM);
}
